// YolactBBoxMask_30236569764338
// MI455X (gfx1250) — compile-verified, matrix operand order checked by reading, not run on hardware
//
#include <hip/hip_runtime.h>
#include <hip/hip_bf16.h>

typedef __attribute__((ext_vector_type(16))) __bf16 v16bf;
typedef __attribute__((ext_vector_type(8)))  __bf16 v8bf;
typedef __attribute__((ext_vector_type(8)))  float  v8f;
typedef unsigned long long u64;
typedef unsigned int u32;

#define BATCH   64
#define NANCH   6402
#define NCLS    80
#define TOPK    200
#define MAXN    100
#define NCOEF   32
#define PROTO_R 6400   // 80*80
#define MSTRIP  128    // rows per block in mask kernel
#define NSTRIPS (PROTO_R / MSTRIP)   // 50

// fast sigmoid: one v_exp + one v_rcp (no IEEE divide expansion)
__device__ __forceinline__ float sigmoidf_(float x) {
    return __builtin_amdgcn_rcpf(1.0f + __expf(-x));
}

// ---------------------------------------------------------------------------
// Kernel A: flatten + softmax(81) + anchors + delta2bbox + coeff flatten
// one thread per (batch, anchor)
// ---------------------------------------------------------------------------
__global__ __launch_bounds__(256)
void k_prep(const float* __restrict__ c0, const float* __restrict__ c1,
            const float* __restrict__ c2, const float* __restrict__ c3,
            const float* __restrict__ c4,
            const float* __restrict__ b0, const float* __restrict__ b1,
            const float* __restrict__ b2, const float* __restrict__ b3,
            const float* __restrict__ b4,
            const float* __restrict__ f0, const float* __restrict__ f1,
            const float* __restrict__ f2, const float* __restrict__ f3,
            const float* __restrict__ f4,
            float* __restrict__ scores, float* __restrict__ bboxes,
            float* __restrict__ coeffs)
{
    int t = blockIdx.x * blockDim.x + threadIdx.x;
    if (t >= BATCH * NANCH) return;
    int b = t / NANCH, a = t % NANCH;

    int loff, f, bs;
    const float *cp, *bp, *fp;
    if (a < 4800)      { loff = 0;    f = 40; bs = 8;   cp = c0; bp = b0; fp = f0; }
    else if (a < 6000) { loff = 4800; f = 20; bs = 16;  cp = c1; bp = b1; fp = f1; }
    else if (a < 6300) { loff = 6000; f = 10; bs = 32;  cp = c2; bp = b2; fp = f2; }
    else if (a < 6375) { loff = 6300; f = 5;  bs = 64;  cp = c3; bp = b3; fp = f3; }
    else               { loff = 6375; f = 3;  bs = 128; cp = c4; bp = b4; fp = f4; }

    int q = a - loff;
    int ai = q % 3, cell = q / 3;
    int y = cell / f, x = cell % f;
    size_t ff  = (size_t)f * f;
    size_t pix = (size_t)y * f + x;

    // ---- softmax over 81 classes, write first 80 class-major ----
    const float* cls = cp + ((size_t)b * 243 + (size_t)ai * 81) * ff + pix;
    float v[81];
    float m = -3.4e38f;
    #pragma unroll 1
    for (int c = 0; c < 81; ++c) { float z = cls[(size_t)c * ff]; v[c] = z; m = fmaxf(m, z); }
    float s = 0.0f;
    #pragma unroll 1
    for (int c = 0; c < 81; ++c) { float e = __expf(v[c] - m); v[c] = e; s += e; }
    float inv = __builtin_amdgcn_rcpf(s);
    #pragma unroll 1
    for (int c = 0; c < NCLS; ++c)
        scores[((size_t)b * NCLS + c) * NANCH + a] = v[c] * inv;

    // ---- anchor (base_size bs, scale 3, ratios {0.5,1,2}) ----
    float r  = (ai == 0) ? 0.5f : ((ai == 1) ? 1.0f : 2.0f);
    float hr = sqrtf(r), wr = 1.0f / hr;
    float aw = bs * wr * 3.0f, ah = bs * hr * 3.0f;
    float acx = bs * 0.5f + (float)x * bs;
    float acy = bs * 0.5f + (float)y * bs;

    // ---- delta2bbox ----
    const float* del = bp + ((size_t)b * 12 + (size_t)ai * 4) * ff + pix;
    float d0 = del[0] * 0.1f;
    float d1 = del[(size_t)1 * ff] * 0.1f;
    float d2 = del[(size_t)2 * ff] * 0.2f;
    float d3 = del[(size_t)3 * ff] * 0.2f;
    const float MAXR = 4.135166556742356f;   // |log(0.016)|
    d2 = fminf(fmaxf(d2, -MAXR), MAXR);
    d3 = fminf(fmaxf(d3, -MAXR), MAXR);
    float gx = acx + aw * d0, gy = acy + ah * d1;
    float gw = aw * __expf(d2), gh = ah * __expf(d3);
    float* ob = bboxes + ((size_t)b * NANCH + a) * 4;
    ob[0] = fminf(fmaxf(gx - 0.5f * gw, 0.0f), 320.0f);
    ob[1] = fminf(fmaxf(gy - 0.5f * gh, 0.0f), 320.0f);
    ob[2] = fminf(fmaxf(gx + 0.5f * gw, 0.0f), 320.0f);
    ob[3] = fminf(fmaxf(gy + 0.5f * gh, 0.0f), 320.0f);

    // ---- coeffs ----
    const float* co = fp + ((size_t)b * 96 + (size_t)ai * 32) * ff + pix;
    float* oc = coeffs + ((size_t)b * NANCH + a) * NCOEF;
    #pragma unroll 1
    for (int k = 0; k < NCOEF; ++k) oc[k] = co[(size_t)k * ff];
}

// ---------------------------------------------------------------------------
// Kernel B: per-(batch,class) descending bitonic sort of 8192 packed keys
// key = score_bits<<32 | (65535 - anchor)  -> stable, lower index wins ties
// ---------------------------------------------------------------------------
__global__ __launch_bounds__(256)
void k_topk(const float* __restrict__ scores,
            u32* __restrict__ topidx, float* __restrict__ topscore)
{
    __shared__ u64 key[8192];
    int bc = blockIdx.x;                 // b*80 + c
    const float* sc = scores + (size_t)bc * NANCH;
    int tid = threadIdx.x;

    for (int i = tid; i < 8192; i += 256) {
        u64 k = 0ull;
        if (i < NANCH)
            k = ((u64)__float_as_uint(sc[i]) << 32) | (u64)(65535u - (u32)i);
        key[i] = k;
    }
    __syncthreads();

    for (int k = 2; k <= 8192; k <<= 1) {
        for (int j = k >> 1; j > 0; j >>= 1) {
            for (int i = tid; i < 8192; i += 256) {
                int ixj = i ^ j;
                if (ixj > i) {
                    u64 a = key[i], bb = key[ixj];
                    bool desc = ((i & k) == 0);
                    if (desc ? (a < bb) : (a > bb)) { key[i] = bb; key[ixj] = a; }
                }
            }
            __syncthreads();
        }
    }
    for (int i = tid; i < TOPK; i += 256) {
        u64 kk = key[i];
        topidx[(size_t)bc * TOPK + i]   = 65535u - (u32)(kk & 0xFFFFFFFFull);
        topscore[(size_t)bc * TOPK + i] = __uint_as_float((u32)(kk >> 32));
    }
}

// ---------------------------------------------------------------------------
// Kernel C: per-(batch,class) fast-NMS: triu IoU column max + keep mask
// ---------------------------------------------------------------------------
__global__ __launch_bounds__(256)
void k_nms(const float* __restrict__ bboxes, const u32* __restrict__ topidx,
           const float* __restrict__ topscore, float* __restrict__ masked)
{
    __shared__ float bx[TOPK * 4];
    __shared__ float ss[TOPK];
    int bc = blockIdx.x;
    int b  = bc / NCLS;
    int tid = threadIdx.x;

    if (tid < TOPK) {
        u32 a = topidx[(size_t)bc * TOPK + tid];
        const float* src = bboxes + ((size_t)b * NANCH + a) * 4;
        bx[tid * 4 + 0] = src[0]; bx[tid * 4 + 1] = src[1];
        bx[tid * 4 + 2] = src[2]; bx[tid * 4 + 3] = src[3];
        ss[tid] = topscore[(size_t)bc * TOPK + tid];
    }
    __syncthreads();

    if (tid < TOPK) {
        float x1 = bx[tid*4], y1 = bx[tid*4+1], x2 = bx[tid*4+2], y2 = bx[tid*4+3];
        float area_i = (x2 - x1) * (y2 - y1);
        float iomax = 0.0f;
        for (int j = 0; j < tid; ++j) {
            float jx1 = bx[j*4], jy1 = bx[j*4+1], jx2 = bx[j*4+2], jy2 = bx[j*4+3];
            float area_j = (jx2 - jx1) * (jy2 - jy1);
            float lx = fmaxf(x1, jx1), ly = fmaxf(y1, jy1);
            float rx = fminf(x2, jx2), ry = fminf(y2, jy2);
            float w = fmaxf(rx - lx, 0.0f), h = fmaxf(ry - ly, 0.0f);
            float ov = w * h;
            float un = fmaxf(area_i + area_j - ov, 1e-6f);
            iomax = fmaxf(iomax, ov / un);
        }
        bool keep = (iomax <= 0.5f) && (ss[tid] > 0.05f);
        masked[(size_t)bc * TOPK + tid] = keep ? ss[tid] : -1.0f;
    }
}

// ---------------------------------------------------------------------------
// Kernel D: per-batch global top-100 of 16000 masked scores (iterative argmax,
// wave32 shuffle reduce + LDS), emit dets/labels and selected coeffs
// ---------------------------------------------------------------------------
__global__ __launch_bounds__(1024)
void k_select(const float* __restrict__ masked, const u32* __restrict__ topidx,
              const float* __restrict__ bboxes, const float* __restrict__ coeffs,
              float* __restrict__ out_dets, float* __restrict__ out_labels,
              float* __restrict__ selcoef)
{
    __shared__ float sc[NCLS * TOPK];   // 16000 floats
    __shared__ u64 wred[32];
    __shared__ int selA[MAXN];
    int b = blockIdx.x;
    int tid = threadIdx.x;

    for (int i = tid; i < NCLS * TOPK; i += 1024)
        sc[i] = masked[(size_t)b * NCLS * TOPK + i];
    __syncthreads();

    for (int it = 0; it < MAXN; ++it) {
        u64 best = 0ull;
        for (int i = tid; i < NCLS * TOPK; i += 1024) {
            // scores in [-1,1]; +2 keeps float bits monotonic; low bits break ties low-index-first
            u64 kkey = ((u64)__float_as_uint(sc[i] + 2.0f) << 32) | (u64)(0xFFFFFFFFu - (u32)i);
            best = best > kkey ? best : kkey;
        }
        #pragma unroll
        for (int off = 16; off > 0; off >>= 1) {
            u64 o = __shfl_xor(best, off, 32);
            best = best > o ? best : o;
        }
        if ((tid & 31) == 0) wred[tid >> 5] = best;
        __syncthreads();
        if (tid == 0) {
            u64 w = wred[0];
            for (int qq = 1; qq < 32; ++qq) w = w > wred[qq] ? w : wred[qq];
            int idx   = (int)(0xFFFFFFFFu - (u32)(w & 0xFFFFFFFFull));
            float val = __uint_as_float((u32)(w >> 32)) - 2.0f;
            sc[idx] = -2.0f;   // below every remaining candidate
            int cls = idx / TOPK, rnk = idx % TOPK;
            u32 a = topidx[((size_t)b * NCLS + cls) * TOPK + rnk];
            const float* bb = bboxes + ((size_t)b * NANCH + a) * 4;
            float* det = out_dets + ((size_t)b * MAXN + it) * 5;
            det[0] = bb[0]; det[1] = bb[1]; det[2] = bb[2]; det[3] = bb[3]; det[4] = val;
            out_labels[(size_t)b * MAXN + it] = (float)cls;
            selA[it] = (int)a;
        }
        __syncthreads();
    }

    for (int e = tid; e < MAXN * NCOEF; e += 1024) {
        int n = e >> 5, k = e & 31;
        selcoef[((size_t)b * MAXN + n) * NCOEF + k] =
            coeffs[((size_t)b * NANCH + selA[n]) * NCOEF + k];
    }
}

// ---------------------------------------------------------------------------
// Kernel E: masks = sigmoid(proto(6400x32) @ cselT(32x100)) via bf16 WMMA.
// One block = (batch, 128-row strip). The 128x32 f32 proto tile is contiguous:
// load once, convert to bf16, stage in 8KB LDS -> proto HBM traffic drops 7x.
// Waves 0..6 each own one n-tile (B kept in VGPRs), sweep 8 m-subtiles reading
// A from LDS (two ds_load_b128 per lane, ISA 16-bit A layout: lanes 0-15 hold
// K {0..7,16..23}, lanes 16-31 hold K {8..15,24..31}; row = lane&15).
// C/D layout: VGPR r, lane l -> M = r + 8*(l>=16), N = l&15.
// ---------------------------------------------------------------------------
__global__ __launch_bounds__(256)
void k_masks(const float* __restrict__ proto, const float* __restrict__ selcoef,
             float* __restrict__ out_masks)
{
    __shared__ __align__(16) __bf16 Atile[MSTRIP * NCOEF];   // 8KB

    int b     = blockIdx.x / NSTRIPS;
    int strip = blockIdx.x % NSTRIPS;
    int tid   = threadIdx.x;
    int wavei = tid >> 5;
    int lane  = tid & 31;

    // ---- stage proto tile: 4096 contiguous floats -> bf16 in LDS ----
    const float* src = proto + ((size_t)b * PROTO_R + (size_t)strip * MSTRIP) * NCOEF;
    #pragma unroll
    for (int e4 = tid; e4 < MSTRIP * NCOEF / 4; e4 += 256) {   // 1024 float4s
        float4 v = ((const float4*)src)[e4];
        __bf16 o[4];
        o[0] = (__bf16)v.x; o[1] = (__bf16)v.y; o[2] = (__bf16)v.z; o[3] = (__bf16)v.w;
        *(uint2*)&Atile[e4 * 4] = *(const uint2*)o;            // 8B LDS store
    }
    __syncthreads();

    if (wavei >= 7) return;          // 7 n-tiles cover 112 >= 100 columns

    int nlo = lane & 15;
    int kb  = (lane >> 4) * 8;       // K groups {kb..kb+7, kb+16..kb+23}
    int n   = wavei * 16 + nlo;      // output mask column (selected detection)

    // ---- B tile: B[k][n] = selcoef[n][k], f32 -> bf16, held for whole block ----
    const float* cf = selcoef + ((size_t)b * MAXN + n) * NCOEF;
    __builtin_prefetch(cf, 0, 0);    // global_prefetch_b8
    v16bf vb;
    #pragma unroll
    for (int i = 0; i < 8; ++i) {
        float lo = (n < MAXN) ? cf[kb + i]      : 0.0f;
        float hi = (n < MAXN) ? cf[kb + 16 + i] : 0.0f;
        vb[i]     = (__bf16)lo;
        vb[8 + i] = (__bf16)hi;
    }

    float* obase = out_masks + (size_t)b * PROTO_R * MAXN;
    #pragma unroll 1
    for (int mt = 0; mt < MSTRIP / 16; ++mt) {               // 8 m-subtiles
        int row_local = mt * 16 + nlo;                        // A row for this lane
        v8bf alo = *(const v8bf*)&Atile[row_local * NCOEF + kb];       // ds_load_b128
        v8bf ahi = *(const v8bf*)&Atile[row_local * NCOEF + kb + 16];  // ds_load_b128
        v16bf va;
        #pragma unroll
        for (int i = 0; i < 8; ++i) { va[i] = alo[i]; va[8 + i] = ahi[i]; }

        v8f c = {};
        c = __builtin_amdgcn_wmma_f32_16x16x32_bf16(
                /*neg_a=*/false, va, /*neg_b=*/false, vb,
                /*c_mod=*/(short)0, c, /*reuse_a=*/false, /*reuse_b=*/false);

        if (n < MAXN) {
            int mbase = strip * MSTRIP + mt * 16 + ((lane >> 4) << 3);
            #pragma unroll
            for (int r = 0; r < 8; ++r)
                obase[(size_t)(mbase + r) * MAXN + n] = sigmoidf_(c[r]);
        }
    }
}

// ---------------------------------------------------------------------------
extern "C" void kernel_launch(void* const* d_in, const int* in_sizes, int n_in,
                              void* d_out, int out_size, void* d_ws, size_t ws_size,
                              hipStream_t stream)
{
    const float* c0 = (const float*)d_in[0];
    const float* c1 = (const float*)d_in[1];
    const float* c2 = (const float*)d_in[2];
    const float* c3 = (const float*)d_in[3];
    const float* c4 = (const float*)d_in[4];
    const float* b0 = (const float*)d_in[5];
    const float* b1 = (const float*)d_in[6];
    const float* b2 = (const float*)d_in[7];
    const float* b3 = (const float*)d_in[8];
    const float* b4 = (const float*)d_in[9];
    const float* f0 = (const float*)d_in[10];
    const float* f1 = (const float*)d_in[11];
    const float* f2 = (const float*)d_in[12];
    const float* f3 = (const float*)d_in[13];
    const float* f4 = (const float*)d_in[14];
    const float* proto = (const float*)d_in[15];

    // workspace layout (~203 MB)
    float* scores   = (float*)d_ws;                                  // 64*80*6402
    float* bboxes   = scores   + (size_t)BATCH * NCLS * NANCH;       // 64*6402*4
    float* coeffs   = bboxes   + (size_t)BATCH * NANCH * 4;          // 64*6402*32
    u32*   topidx   = (u32*)(coeffs + (size_t)BATCH * NANCH * NCOEF);// 64*80*200
    float* topscore = (float*)(topidx + (size_t)BATCH * NCLS * TOPK);// 64*80*200
    float* maskedsc = topscore + (size_t)BATCH * NCLS * TOPK;        // 64*80*200
    float* selcoef  = maskedsc + (size_t)BATCH * NCLS * TOPK;        // 64*100*32

    // output: dets (64,100,5) || labels (64,100) || masks (64,80,80,100)
    float* out_dets   = (float*)d_out;
    float* out_labels = out_dets + (size_t)BATCH * MAXN * 5;
    float* out_masks  = out_labels + (size_t)BATCH * MAXN;

    int prep_blocks = (BATCH * NANCH + 255) / 256;
    k_prep<<<prep_blocks, 256, 0, stream>>>(c0, c1, c2, c3, c4,
                                            b0, b1, b2, b3, b4,
                                            f0, f1, f2, f3, f4,
                                            scores, bboxes, coeffs);
    k_topk<<<BATCH * NCLS, 256, 0, stream>>>(scores, topidx, topscore);
    k_nms<<<BATCH * NCLS, 256, 0, stream>>>(bboxes, topidx, topscore, maskedsc);
    k_select<<<BATCH, 1024, 0, stream>>>(maskedsc, topidx, bboxes, coeffs,
                                         out_dets, out_labels, selcoef);
    // one block per (batch, 128-row strip): 64 * 50 = 3200 blocks
    k_masks<<<BATCH * NSTRIPS, 256, 0, stream>>>(proto, selcoef, out_masks);
}
